// SegBlock_16587163697762
// MI455X (gfx1250) — matrix operand order repaired and checked by reading, not compiled, not run on hardware
//
#include <hip/hip_runtime.h>

typedef __attribute__((ext_vector_type(16))) __bf16 v16bf;
typedef __attribute__((ext_vector_type(2)))  __bf16 v2bf;
typedef __attribute__((ext_vector_type(2)))  float  v2f;
typedef __attribute__((ext_vector_type(8)))  float  v8f;

#define GRID_G 300
#define NCHUNK 10     // ceil(300/32)
#define LSTR   36     // LDS row stride in uints: 16B-aligned rows AND bank-conflict-free b128 reads

// float pair -> packed bf16 (RNE via fptrunc; backend picks native cvt if present)
__device__ __forceinline__ unsigned pack_bf16(float a, float b) {
    v2f f; f.x = a; f.y = b;
    v2bf r = __builtin_convertvector(f, v2bf);
    return __builtin_bit_cast(unsigned, r);
}

// ---------------- LayerNorm over D=128, row-major out ----------------
__global__ __launch_bounds__(128) void ln1_kernel(
    const float* __restrict__ x, const float* __restrict__ w,
    const float* __restrict__ b, float* __restrict__ y)
{
    const int row = blockIdx.x, t = threadIdx.x;
    float v = x[(size_t)row * 128 + t];
    float s = v, q = v * v;
    #pragma unroll
    for (int o = 16; o; o >>= 1) { s += __shfl_xor(s, o, 32); q += __shfl_xor(q, o, 32); }
    __shared__ float ls[4], lq[4];
    if ((t & 31) == 0) { ls[t >> 5] = s; lq[t >> 5] = q; }
    __syncthreads();
    s = ls[0] + ls[1] + ls[2] + ls[3];
    q = lq[0] + lq[1] + lq[2] + lq[3];
    float mu = s * (1.0f / 128.0f);
    float var = q * (1.0f / 128.0f) - mu * mu;
    float inv = rsqrtf(var + 1e-5f);
    y[(size_t)row * 128 + t] = (v - mu) * inv * w[t] + b[t];
}

// ------------- LayerNorm over 256 + transpose to [B,256,196] -------------
__global__ __launch_bounds__(256) void ln2t_kernel(
    const float* __restrict__ h, const float* __restrict__ w,
    const float* __restrict__ b, float* __restrict__ y)
{
    const int rn = blockIdx.x, t = threadIdx.x;   // rn = b*196 + n
    float v = h[(size_t)rn * 256 + t];
    float s = v, q = v * v;
    #pragma unroll
    for (int o = 16; o; o >>= 1) { s += __shfl_xor(s, o, 32); q += __shfl_xor(q, o, 32); }
    __shared__ float ls[8], lq[8];
    if ((t & 31) == 0) { ls[t >> 5] = s; lq[t >> 5] = q; }
    __syncthreads();
    s = 0.f; q = 0.f;
    #pragma unroll
    for (int k = 0; k < 8; ++k) { s += ls[k]; q += lq[k]; }
    float mu = s * (1.0f / 256.0f);
    float var = q * (1.0f / 256.0f) - mu * mu;
    float inv = rsqrtf(var + 1e-5f);
    float o = (v - mu) * inv * w[t] + b[t];
    int bb = rn / 196, n = rn - bb * 196;
    y[((size_t)(bb * 256 + t)) * 196 + n] = o;
}

// ---------------- Fused FourierKAN GEMM (K-split partial) ----------------
// P[z][m, o] = sum_{i in split z} sum_g cos(Xn[m,i]*(g+1))*C0[o,i,g] + sin(..)*C1[o,i,g]
// BM=64, BN=64, 256 threads = 8 waves (4M x 2N), each wave 16x32 via 2 accs.
__global__ __launch_bounds__(256) void fkan_gemm(
    const float* __restrict__ Xn, const float* __restrict__ C0,
    const float* __restrict__ C1, float* __restrict__ P,
    int M, int I, int O, int isplit)
{
    __shared__ __attribute__((aligned(16))) unsigned lAc[64 * LSTR];
    __shared__ __attribute__((aligned(16))) unsigned lAs[64 * LSTR];
    __shared__ __attribute__((aligned(16))) unsigned lB0[64 * LSTR];
    __shared__ __attribute__((aligned(16))) unsigned lB1[64 * LSTR];

    const int tid  = threadIdx.x;
    const int lane = tid & 31;
    const int wave = tid >> 5;
    const int wm   = wave >> 1;          // 0..3  (M sub-tile of 16)
    const int wn   = wave & 1;           // 0..1  (N sub-tile of 32)
    const int m0   = blockIdx.x * 64;
    const int n0   = blockIdx.y * 64;
    const int i0   = blockIdx.z * isplit;
    const int iend = (i0 + isplit < I) ? (i0 + isplit) : I;
    float* Pk = P + (size_t)blockIdx.z * M * O;

    // A-generation role: 64 rows x 32 g, 8 g per thread
    const int ra = tid >> 2;             // 0..63
    const int ga = (tid & 3) * 8;        // start g within chunk
    // B-load role: 64 cols x 32 g (x2 matrices), 8 floats per thread each
    const int nb = tid >> 2;             // 0..63
    const int eb = (tid & 3) * 8;

    const int rowA = m0 + ra;
    // clamp instead of predicating: OOB rows compute garbage that is never stored
    const int rowL = (rowA < M) ? rowA : (M - 1);
    const bool oOK = (n0 + nb) < O;
    const float* __restrict__ xrow = Xn + (size_t)rowL * I;
    const size_t cbase = ((size_t)(n0 + nb) * I) * GRID_G;

    v8f acc0 = {0.f,0.f,0.f,0.f,0.f,0.f,0.f,0.f};
    v8f acc1 = {0.f,0.f,0.f,0.f,0.f,0.f,0.f,0.f};

    const int arow  = (wm * 16 + (lane & 15)) * LSTR + ((lane & 16) ? 8 : 0);
    const int brow0 = (wn * 32 + (lane & 15)) * LSTR + ((lane >> 4) << 3);
    const int brow1 = brow0 + 16 * LSTR;
    const int ai = ra * LSTR + (ga >> 1);
    const int bi = nb * LSTR + (eb >> 1);

    for (int i = i0; i < iend; ++i) {
        const float xvi = xrow[i];
        float s1, c1v, s32, c32, sst, cst;
        __sincosf(xvi, &s1, &c1v);                       // per-g rotation step
        __sincosf(32.0f * xvi, &s32, &c32);              // chunk-to-chunk step
        __sincosf(xvi * (float)(ga + 1), &sst, &cst);    // phase at first slot
        const size_t crow = cbase + (size_t)i * GRID_G;

        for (int c = 0; c < NCHUNK; ++c) {
            const int g0 = c * 32;
            // ---- generate A tiles (cos & sin) via angle-addition recurrence ----
            {
                float co = cst, si = sst;
                uint4 pc, ps;
                unsigned* pcu = &pc.x;
                unsigned* psu = &ps.x;
                #pragma unroll
                for (int j = 0; j < 4; ++j) {
                    float c2 = co * c1v - si * s1;
                    float sx = si * c1v + co * s1;
                    pcu[j] = pack_bf16(co, c2);
                    psu[j] = pack_bf16(si, sx);
                    co = c2 * c1v - sx * s1;             // advance to next pair
                    si = sx * c1v + c2 * s1;
                }
                *(uint4*)&lAc[ai] = pc;
                *(uint4*)&lAs[ai] = ps;
                // advance chunk start by 32 steps
                float cn = cst * c32 - sst * s32;
                sst = sst * c32 + cst * s32;
                cst = cn;
            }
            // ---- load B tiles (fp32 coeffs -> bf16) ----
            {
                const int ge = g0 + eb;
                uint4 b0, b1;
                if (oOK && (ge + 7 < GRID_G)) {
                    float4 f0 = *(const float4*)(C0 + crow + ge);
                    float4 f1 = *(const float4*)(C0 + crow + ge + 4);
                    b0.x = pack_bf16(f0.x, f0.y); b0.y = pack_bf16(f0.z, f0.w);
                    b0.z = pack_bf16(f1.x, f1.y); b0.w = pack_bf16(f1.z, f1.w);
                    f0 = *(const float4*)(C1 + crow + ge);
                    f1 = *(const float4*)(C1 + crow + ge + 4);
                    b1.x = pack_bf16(f0.x, f0.y); b1.y = pack_bf16(f0.z, f0.w);
                    b1.z = pack_bf16(f1.x, f1.y); b1.w = pack_bf16(f1.z, f1.w);
                } else {
                    float t0[8], t1[8];
                    #pragma unroll
                    for (int e = 0; e < 8; ++e) {
                        const bool v = oOK && (ge + e < GRID_G);
                        t0[e] = v ? C0[crow + ge + e] : 0.0f;
                        t1[e] = v ? C1[crow + ge + e] : 0.0f;
                    }
                    b0.x = pack_bf16(t0[0], t0[1]); b0.y = pack_bf16(t0[2], t0[3]);
                    b0.z = pack_bf16(t0[4], t0[5]); b0.w = pack_bf16(t0[6], t0[7]);
                    b1.x = pack_bf16(t1[0], t1[1]); b1.y = pack_bf16(t1[2], t1[3]);
                    b1.z = pack_bf16(t1[4], t1[5]); b1.w = pack_bf16(t1[6], t1[7]);
                }
                *(uint4*)&lB0[bi] = b0;
                *(uint4*)&lB1[bi] = b1;
            }
            __syncthreads();
            // ---- WMMA: cos part against C0, sin part against C1 ----
            {
                union FU { uint4 q[2]; v16bf v; } fa, fb0, fb1;
                fa.q[0]  = *(const uint4*)&lAc[arow];
                fa.q[1]  = *(const uint4*)&lAc[arow + 4];
                fb0.q[0] = *(const uint4*)&lB0[brow0];
                fb0.q[1] = *(const uint4*)&lB0[brow0 + 4];
                fb1.q[0] = *(const uint4*)&lB0[brow1];
                fb1.q[1] = *(const uint4*)&lB0[brow1 + 4];
                acc0 = __builtin_amdgcn_wmma_f32_16x16x32_bf16(false, fa.v, false, fb0.v,
                                                               (short)0, acc0, false, false);
                acc1 = __builtin_amdgcn_wmma_f32_16x16x32_bf16(false, fa.v, false, fb1.v,
                                                               (short)0, acc1, false, false);
                fa.q[0]  = *(const uint4*)&lAs[arow];
                fa.q[1]  = *(const uint4*)&lAs[arow + 4];
                fb0.q[0] = *(const uint4*)&lB1[brow0];
                fb0.q[1] = *(const uint4*)&lB1[brow0 + 4];
                fb1.q[0] = *(const uint4*)&lB1[brow1];
                fb1.q[1] = *(const uint4*)&lB1[brow1 + 4];
                acc0 = __builtin_amdgcn_wmma_f32_16x16x32_bf16(false, fa.v, false, fb0.v,
                                                               (short)0, acc0, false, false);
                acc1 = __builtin_amdgcn_wmma_f32_16x16x32_bf16(false, fa.v, false, fb1.v,
                                                               (short)0, acc1, false, false);
            }
            __syncthreads();
        }
    }

    // ---- store partials (no bias; ISA 32-bit C/D 16x16 layout) ----
    const int mbase = m0 + wm * 16 + ((lane >> 4) << 3);
    #pragma unroll
    for (int sub = 0; sub < 2; ++sub) {
        const v8f& a = sub ? acc1 : acc0;
        const int col = n0 + wn * 32 + sub * 16 + (lane & 15);
        if (col < O) {
            #pragma unroll
            for (int v = 0; v < 8; ++v) {
                const int m = mbase + v;
                if (m < M) Pk[(size_t)m * O + col] = a[v];
            }
        }
    }
}

// ---- reduce 4 K-split partials + bias -> h  (fc1, O=256) ----
__global__ __launch_bounds__(256) void reduce1_kernel(
    const float* __restrict__ P, const float* __restrict__ bias, float* __restrict__ Y)
{
    const int idx = blockIdx.x * 256 + threadIdx.x;          // < 1568*256
    const size_t MO = (size_t)1568 * 256;
    float s = bias[idx & 255];
    s += P[idx]; s += P[MO + idx]; s += P[2 * MO + idx]; s += P[3 * MO + idx];
    Y[idx] = s;
}

// ---- reduce fc2 partials + bias, transpose back to [B,N,2D] ----
__global__ __launch_bounds__(256) void reduce2_kernel(
    const float* __restrict__ P, const float* __restrict__ bias, float* __restrict__ out)
{
    const int idx = blockIdx.x * 256 + threadIdx.x;          // < 2048*196
    const int m = idx / 196, j = idx - m * 196;              // m = b*256 + ch
    const size_t MO = (size_t)2048 * 196;
    float s = bias[j];
    s += P[idx]; s += P[MO + idx]; s += P[2 * MO + idx]; s += P[3 * MO + idx];
    out[((size_t)((m >> 8) * 196 + j)) * 256 + (m & 255)] = s;
}

extern "C" void kernel_launch(void* const* d_in, const int* in_sizes, int n_in,
                              void* d_out, int out_size, void* d_ws, size_t ws_size,
                              hipStream_t stream) {
    const float* x     = (const float*)d_in[0];
    const float* ln1w  = (const float*)d_in[1];
    const float* ln1b  = (const float*)d_in[2];
    const float* fc1c  = (const float*)d_in[3];
    const float* fc1bi = (const float*)d_in[4];
    const float* ln2w  = (const float*)d_in[5];
    const float* ln2b  = (const float*)d_in[6];
    const float* fc2c  = (const float*)d_in[7];
    const float* fc2bi = (const float*)d_in[8];
    float* out = (float*)d_out;

    float* ws  = (float*)d_ws;
    float* xn1 = ws;                       // 1568*128
    float* h   = xn1 + 1568 * 128;         // 1568*256
    float* xn2 = h + 1568 * 256;           // 2048*196
    float* P1  = xn2 + 2048 * 196;         // 4 * 1568*256
    float* P2  = P1 + (size_t)4 * 1568 * 256;  // 4 * 2048*196

    const size_t FC1_C1 = (size_t)256 * 128 * GRID_G;   // cos/sin halves of [2,O,I,G]
    const size_t FC2_C1 = (size_t)196 * 196 * GRID_G;

    ln1_kernel<<<1568, 128, 0, stream>>>(x, ln1w, ln1b, xn1);
    fkan_gemm<<<dim3(25, 4, 4), 256, 0, stream>>>(xn1, fc1c, fc1c + FC1_C1, P1,
                                                  1568, 128, 256, 32);
    reduce1_kernel<<<1568, 256, 0, stream>>>(P1, fc1bi, h);
    ln2t_kernel<<<1568, 256, 0, stream>>>(h, ln2w, ln2b, xn2);
    fkan_gemm<<<dim3(32, 4, 4), 256, 0, stream>>>(xn2, fc2c, fc2c + FC2_C1, P2,
                                                  2048, 196, 196, 49);
    reduce2_kernel<<<1568, 256, 0, stream>>>(P2, fc2bi, out);
}
